// MultiheadAttention_10849087389767
// MI455X (gfx1250) — matrix operand order repaired and checked by reading, not compiled, not run on hardware
//
#include <hip/hip_runtime.h>
#include <math.h>

// Problem constants (from reference)
#define Bc  2
#define Sc  2048
#define Dc  2048
#define Hc  16
#define HDc 128
#define BSc (Bc * Sc)          // 4096 tokens
#define LN10000 9.2103403719761836f
#define SCALE 0.022097086912079608f  // 1/sqrt(2048)
#define NEGINF (-1.0e30f)

typedef __attribute__((ext_vector_type(16))) __bf16 v16bf;
typedef __attribute__((ext_vector_type(8)))  __bf16 v8bf;
typedef __attribute__((ext_vector_type(8)))  float  v8f;

// ---------------------------------------------------------------------------
// WMMA helper: D = A(16x32 bf16) * B(32x16 bf16) + C(16x16 f32)
// ---------------------------------------------------------------------------
__device__ __forceinline__ v8f wmma_bf16(v16bf a, v16bf b, v8f c) {
  return __builtin_amdgcn_wmma_f32_16x16x32_bf16(
      /*neg_a=*/false, a, /*neg_b=*/false, b,
      /*c_mod=*/(short)0, c, /*reuse_a=*/false, /*reuse_b=*/false);
}

// A fragment from a lane-resolved pointer (p = &A[row*lda + khalf*8]):
// two 16B loads at +0 and +16 elements.
__device__ __forceinline__ v16bf ld_fragA(const __bf16* p) {
  v8bf lo = *(const v8bf*)(p);
  v8bf hi = *(const v8bf*)(p + 16);
  v16bf r;
#pragma unroll
  for (int i = 0; i < 8; ++i) { r[i] = lo[i]; r[i + 8] = hi[i]; }
  return r;
}

// B fragment from a lane-resolved pointer (p = &BT[col*ldb + khalf*8]):
// two 16B loads at +0 and +16 elements.
__device__ __forceinline__ v16bf ld_fragB(const __bf16* p) {
  v8bf lo = *(const v8bf*)(p);
  v8bf hi = *(const v8bf*)(p + 16);
  v16bf r;
#pragma unroll
  for (int i = 0; i < 8; ++i) { r[i] = lo[i]; r[i + 8] = hi[i]; }
  return r;
}

// Indexed wrappers (used by the flash kernel)
__device__ __forceinline__ v16bf load_a_frag(const __bf16* A, int lda, int m0,
                                             int k0, int lane) {
  return ld_fragA(A + (size_t)(m0 + (lane & 15)) * lda + k0 + (lane >> 4) * 8);
}
__device__ __forceinline__ v16bf load_b_frag(const __bf16* BT, int ldb, int n0,
                                             int k0, int lane) {
  return ld_fragB(BT + (size_t)(n0 + (lane & 15)) * ldb + k0 + (lane >> 4) * 8);
}

// A fragment from a 16x32 bf16 tile resident in LDS (row stride 32)
__device__ __forceinline__ v16bf load_a_frag_lds(const __bf16* P, int lane) {
  return ld_fragA(P + (lane & 15) * 32 + (lane >> 4) * 8);
}

// Async copy of one contiguous 8 KB tile (32 keys x 128 hd bf16) global -> LDS.
// 16 per-lane b128 async transfers; tracked by ASYNCcnt.
__device__ __forceinline__ void stage_tile_async(const __bf16* g, const __bf16* l,
                                                 int lane) {
  unsigned lbase = (unsigned)(size_t)l + (unsigned)lane * 16u;
  const char* gb = (const char*)g + lane * 16;
#pragma unroll
  for (int c = 0; c < 16; ++c) {
    unsigned lo = lbase + (unsigned)c * 512u;  // 32 lanes * 16B per sweep
    const char* gp = gb + c * 512;
    asm volatile("global_load_async_to_lds_b128 %0, %1, off"
                 :: "v"(lo), "v"(gp) : "memory");
  }
}

// ---------------------------------------------------------------------------
// Kernel 1: fp32 -> bf16 elementwise
// ---------------------------------------------------------------------------
__global__ void cvt_bf16_kernel(const float* __restrict__ in,
                                __bf16* __restrict__ out, int n) {
  for (int i = blockIdx.x * blockDim.x + threadIdx.x; i < n;
       i += gridDim.x * blockDim.x)
    out[i] = (__bf16)in[i];
}

// ---------------------------------------------------------------------------
// Kernel 2: transpose + convert a DxD fp32 weight into bf16 W^T
// ---------------------------------------------------------------------------
__global__ __launch_bounds__(1024) void tcvt_kernel(const float* __restrict__ in,
                                                    __bf16* __restrict__ out) {
  __shared__ float t[32][33];
  int x = blockIdx.x * 32 + threadIdx.x;
  int y = blockIdx.y * 32 + threadIdx.y;
  t[threadIdx.y][threadIdx.x] = in[(size_t)y * Dc + x];
  __syncthreads();
  int xo = blockIdx.y * 32 + threadIdx.x;
  int yo = blockIdx.x * 32 + threadIdx.y;
  out[(size_t)yo * Dc + xo] = (__bf16)t[threadIdx.x][threadIdx.y];
}

// ---------------------------------------------------------------------------
// Pipelined 32x64-per-wave GEMM core: ping-pong unroll over K (step 64) with
// incrementally-advanced lane pointers; no register copy-rotation.
// acc[2][4] accumulates C tiles for rows {m0, m0+16} x cols {n0 + 16j}.
// ---------------------------------------------------------------------------
__device__ __forceinline__ void gemm_core_32x64(
    const __bf16* __restrict__ A, const __bf16* __restrict__ BT,
    int m0, int n0, int lane, v8f acc[2][4]) {
  const int row = lane & 15;
  const int khA = (lane >> 4) * 8;
  const int khB = (lane >> 4) * 8;
  const __bf16* pa0 = A + (size_t)(m0 + row) * Dc + khA;
  const __bf16* pa1 = A + (size_t)(m0 + 16 + row) * Dc + khA;
  const __bf16* pb0 = BT + (size_t)(n0 + row) * Dc + khB;
  const __bf16* pb1 = pb0 + (size_t)16 * Dc;
  const __bf16* pb2 = pb0 + (size_t)32 * Dc;
  const __bf16* pb3 = pb0 + (size_t)48 * Dc;

  // phase-0 fragments at k=0
  v16bf a0c = ld_fragA(pa0), a1c = ld_fragA(pa1);
  v16bf bc0 = ld_fragB(pb0), bc1 = ld_fragB(pb1);
  v16bf bc2 = ld_fragB(pb2), bc3 = ld_fragB(pb3);

  for (int it = 0; it < (Dc / 64) - 1; ++it) {
    // phase-1 fragments at k+32 (immediate offset +64B)
    v16bf a0n = ld_fragA(pa0 + 32), a1n = ld_fragA(pa1 + 32);
    v16bf bn0 = ld_fragB(pb0 + 32), bn1 = ld_fragB(pb1 + 32);
    v16bf bn2 = ld_fragB(pb2 + 32), bn3 = ld_fragB(pb3 + 32);
    acc[0][0] = wmma_bf16(a0c, bc0, acc[0][0]);
    acc[1][0] = wmma_bf16(a1c, bc0, acc[1][0]);
    acc[0][1] = wmma_bf16(a0c, bc1, acc[0][1]);
    acc[1][1] = wmma_bf16(a1c, bc1, acc[1][1]);
    acc[0][2] = wmma_bf16(a0c, bc2, acc[0][2]);
    acc[1][2] = wmma_bf16(a1c, bc2, acc[1][2]);
    acc[0][3] = wmma_bf16(a0c, bc3, acc[0][3]);
    acc[1][3] = wmma_bf16(a1c, bc3, acc[1][3]);
    // phase-0 fragments at k+64 (immediate offset +128B), fresh definitions
    a0c = ld_fragA(pa0 + 64); a1c = ld_fragA(pa1 + 64);
    bc0 = ld_fragB(pb0 + 64); bc1 = ld_fragB(pb1 + 64);
    bc2 = ld_fragB(pb2 + 64); bc3 = ld_fragB(pb3 + 64);
    acc[0][0] = wmma_bf16(a0n, bn0, acc[0][0]);
    acc[1][0] = wmma_bf16(a1n, bn0, acc[1][0]);
    acc[0][1] = wmma_bf16(a0n, bn1, acc[0][1]);
    acc[1][1] = wmma_bf16(a1n, bn1, acc[1][1]);
    acc[0][2] = wmma_bf16(a0n, bn2, acc[0][2]);
    acc[1][2] = wmma_bf16(a1n, bn2, acc[1][2]);
    acc[0][3] = wmma_bf16(a0n, bn3, acc[0][3]);
    acc[1][3] = wmma_bf16(a1n, bn3, acc[1][3]);
    pa0 += 64; pa1 += 64;
    pb0 += 64; pb1 += 64; pb2 += 64; pb3 += 64;
  }
  // tail: phase-0 holds k = Dc-64; load phase-1 at Dc-32 and finish both
  {
    v16bf a0n = ld_fragA(pa0 + 32), a1n = ld_fragA(pa1 + 32);
    v16bf bn0 = ld_fragB(pb0 + 32), bn1 = ld_fragB(pb1 + 32);
    v16bf bn2 = ld_fragB(pb2 + 32), bn3 = ld_fragB(pb3 + 32);
    acc[0][0] = wmma_bf16(a0c, bc0, acc[0][0]);
    acc[1][0] = wmma_bf16(a1c, bc0, acc[1][0]);
    acc[0][1] = wmma_bf16(a0c, bc1, acc[0][1]);
    acc[1][1] = wmma_bf16(a1c, bc1, acc[1][1]);
    acc[0][2] = wmma_bf16(a0c, bc2, acc[0][2]);
    acc[1][2] = wmma_bf16(a1c, bc2, acc[1][2]);
    acc[0][3] = wmma_bf16(a0c, bc3, acc[0][3]);
    acc[1][3] = wmma_bf16(a1c, bc3, acc[1][3]);
    acc[0][0] = wmma_bf16(a0n, bn0, acc[0][0]);
    acc[1][0] = wmma_bf16(a1n, bn0, acc[1][0]);
    acc[0][1] = wmma_bf16(a0n, bn1, acc[0][1]);
    acc[1][1] = wmma_bf16(a1n, bn1, acc[1][1]);
    acc[0][2] = wmma_bf16(a0n, bn2, acc[0][2]);
    acc[1][2] = wmma_bf16(a1n, bn2, acc[1][2]);
    acc[0][3] = wmma_bf16(a0n, bn3, acc[0][3]);
    acc[1][3] = wmma_bf16(a1n, bn3, acc[1][3]);
  }
}

// ---------------------------------------------------------------------------
// Kernel 3: fused QKV GEMM + RoPE. grid = (BS/128, D/64, 3); block = 128 (4 waves)
// Q,K written [b,h,s,hd] bf16 (post-RoPE); V written transposed [b,h,hd,s].
// ---------------------------------------------------------------------------
__global__ __launch_bounds__(128) void qkv_rope_kernel(
    const __bf16* __restrict__ xb,  const __bf16* __restrict__ wqT,
    const __bf16* __restrict__ wkT, const __bf16* __restrict__ wvT,
    __bf16* __restrict__ Qb, __bf16* __restrict__ Kb, __bf16* __restrict__ VbT) {
  const int lane = threadIdx.x & 31;
  const int w    = threadIdx.x >> 5;
  const int z    = blockIdx.z;
  const __bf16* WT = (z == 0) ? wqT : (z == 1) ? wkT : wvT;
  const int m0 = blockIdx.x * 128 + w * 32;
  const int n0 = blockIdx.y * 64;

  v8f acc[2][4] = {};
  gemm_core_32x64(xb, WT, m0, n0, lane, acc);

  const int g  = lane >> 4;
  const int nl = lane & 15;
#pragma unroll
  for (int mi = 0; mi < 2; ++mi) {
#pragma unroll
    for (int j = 0; j < 4; ++j) {
      int d  = n0 + j * 16 + nl;
      int h  = d >> 7;       // /HDc
      int hd = d & 127;
      float invf = __expf(-(float)(d & ~1) * (LN10000 / (float)Dc));
#pragma unroll
      for (int r = 0; r < 8; ++r) {
        int row  = m0 + mi * 16 + g * 8 + r;
        int bidx = row >> 11;  // /Sc
        int pos  = row & 2047;
        float v = acc[mi][j][r];
        if (z < 2) {  // RoPE on Q,K only; complex partner is in adjacent lane
          float pv = __shfl_xor(v, 1, 32);
          float ang = (float)pos * invf;
          float c = __cosf(ang), s = __sinf(ang);
          v = ((d & 1) == 0) ? (v * c - pv * s) : (pv * s + v * c);
        }
        __bf16 o = (__bf16)v;
        if (z == 2) {
          VbT[(((size_t)(bidx * Hc + h) * HDc + hd) * Sc) + pos] = o;
        } else {
          __bf16* dst = (z == 0) ? Qb : Kb;
          dst[(((size_t)(bidx * Hc + h) * Sc + pos) * HDc) + hd] = o;
        }
      }
    }
  }
}

// ---------------------------------------------------------------------------
// Kernel 4: flash attention, one wave per 16-row q tile. grid = B*H*(S/16).
// K tiles are staged global->LDS with async b128 copies in a 2-deep ring
// (ASYNCcnt), prefetching step st+1 while WMMAs of step st execute.
// ---------------------------------------------------------------------------
__global__ __launch_bounds__(32) void flash_attn_kernel(
    const __bf16* __restrict__ Qb, const __bf16* __restrict__ Kb,
    const __bf16* __restrict__ VbT, __bf16* __restrict__ Ab) {
  __shared__ __align__(16) __bf16 Ksh[2][32][HDc];  // 2 x 8 KB K-tile ring
  __shared__ __align__(16) __bf16 Pb[16 * 32];      // P staging for A-frag relayout
  const int lane = threadIdx.x & 31;
  const int qt = blockIdx.x & (Sc / 16 - 1);
  const int bh = blockIdx.x / (Sc / 16);
  const int b = bh / Hc, h = bh % Hc;
  const __bf16* Qp = Qb + (size_t)bh * Sc * HDc;
  const __bf16* Kp = Kb + (size_t)bh * Sc * HDc;
  const __bf16* Vp = VbT + (size_t)bh * HDc * Sc;
  const int q0 = qt * 16;
  const int g = lane >> 4, nl = lane & 15;

  // Q A-fragments for the whole HD=128 contraction, kept in registers
  v16bf qa[4];
#pragma unroll
  for (int kk = 0; kk < 4; ++kk) qa[kk] = load_a_frag(Qp, HDc, q0, kk * 32, lane);

  v8f o[8] = {};                 // 16 x 128 output accumulator (8 C-tiles)
  float m_i[8], l_i[8];
#pragma unroll
  for (int r = 0; r < 8; ++r) { m_i[r] = NEGINF; l_i[r] = 0.0f; }

  const int nsteps = (q0 + 16 + 31) >> 5;  // 32-key steps up to the diagonal

  // preload K tile for step 0
  stage_tile_async(Kp, &Ksh[0][0][0], lane);

  for (int st = 0; st < nsteps; ++st) {
    const int kb = st * 32;
    const int buf = st & 1;
    if (st + 1 < nsteps) {
      // prior iteration's ds reads of Ksh[buf^1] must retire before overwrite
      asm volatile("s_wait_dscnt 0x0" ::: "memory");
      stage_tile_async(Kp + (size_t)(kb + 32) * HDc, &Ksh[buf ^ 1][0][0], lane);
      // 16 newest (next tile) may stay in flight; current tile is complete
      asm volatile("s_wait_asynccnt 0x10" ::: "memory");
    } else {
      asm volatile("s_wait_asynccnt 0x0" ::: "memory");
    }
    const __bf16* kt = &Ksh[buf][0][0];

    // scores: two 16x16 subtiles, contract over HD (B-frags from LDS)
    v8f s0 = {}, s1 = {};
#pragma unroll
    for (int kk = 0; kk < 4; ++kk) {
      v16bf b0 = load_b_frag(kt, HDc, 0,  kk * 32, lane);
      s0 = wmma_bf16(qa[kk], b0, s0);
      v16bf b1 = load_b_frag(kt, HDc, 16, kk * 32, lane);
      s1 = wmma_bf16(qa[kk], b1, s1);
    }

    // scale, causal mask, online softmax
    float alpha[8];
#pragma unroll
    for (int r = 0; r < 8; ++r) {
      int qi = q0 + g * 8 + r;
      float v0 = s0[r] * SCALE;
      float v1 = s1[r] * SCALE;
      if (kb + nl      > qi) v0 = NEGINF;
      if (kb + 16 + nl > qi) v1 = NEGINF;
      float mx = fmaxf(v0, v1);
#pragma unroll
      for (int off = 8; off >= 1; off >>= 1) mx = fmaxf(mx, __shfl_xor(mx, off, 32));
      float mn = fmaxf(m_i[r], mx);
      alpha[r] = __expf(m_i[r] - mn);
      float p0 = __expf(v0 - mn);
      float p1 = __expf(v1 - mn);
      s0[r] = p0; s1[r] = p1;
      float rs = p0 + p1;
#pragma unroll
      for (int off = 8; off >= 1; off >>= 1) rs += __shfl_xor(rs, off, 32);
      l_i[r] = l_i[r] * alpha[r] + rs;
      m_i[r] = mn;
    }

    // stage P (16x32 bf16) through LDS to re-layout into an A fragment
#pragma unroll
    for (int r = 0; r < 8; ++r) {
      Pb[(g * 8 + r) * 32 + nl]      = (__bf16)s0[r];
      Pb[(g * 8 + r) * 32 + 16 + nl] = (__bf16)s1[r];
    }
    asm volatile("s_wait_dscnt 0x0" ::: "memory");
    v16bf pa = load_a_frag_lds(Pb, lane);

    // O = O*alpha + P @ V  (8 hd chunks; V^T rows contiguous in key)
#pragma unroll
    for (int c = 0; c < 8; ++c) {
#pragma unroll
      for (int r = 0; r < 8; ++r) o[c][r] *= alpha[r];
      v16bf vb = load_b_frag(Vp, Sc, c * 16, kb, lane);
      o[c] = wmma_bf16(pa, vb, o[c]);
    }
  }

  // normalize and store bf16 attention output row-major [token, D]
#pragma unroll
  for (int r = 0; r < 8; ++r) l_i[r] = 1.0f / l_i[r];
#pragma unroll
  for (int c = 0; c < 8; ++c) {
    int d = h * HDc + c * 16 + nl;
#pragma unroll
    for (int r = 0; r < 8; ++r) {
      int tok = b * Sc + q0 + g * 8 + r;
      Ab[(size_t)tok * Dc + d] = (__bf16)(o[c][r] * l_i[r]);
    }
  }
}

// ---------------------------------------------------------------------------
// Kernel 5: output projection GEMM -> fp32 d_out. grid = (BS/128, D/64), 128 thr.
// ---------------------------------------------------------------------------
__global__ __launch_bounds__(128) void out_proj_kernel(
    const __bf16* __restrict__ Ab, const __bf16* __restrict__ woT,
    float* __restrict__ out) {
  const int lane = threadIdx.x & 31;
  const int w    = threadIdx.x >> 5;
  const int m0 = blockIdx.x * 128 + w * 32;
  const int n0 = blockIdx.y * 64;

  v8f acc[2][4] = {};
  gemm_core_32x64(Ab, woT, m0, n0, lane, acc);

  const int g = lane >> 4, nl = lane & 15;
#pragma unroll
  for (int mi = 0; mi < 2; ++mi) {
#pragma unroll
    for (int j = 0; j < 4; ++j) {
      int d = n0 + j * 16 + nl;
#pragma unroll
      for (int r = 0; r < 8; ++r) {
        int row = m0 + mi * 16 + g * 8 + r;
        out[(size_t)row * Dc + d] = acc[mi][j][r];
      }
    }
  }
}

// ---------------------------------------------------------------------------
// Launcher
// ---------------------------------------------------------------------------
extern "C" void kernel_launch(void* const* d_in, const int* in_sizes, int n_in,
                              void* d_out, int out_size, void* d_ws, size_t ws_size,
                              hipStream_t stream) {
  const float* x  = (const float*)d_in[0];
  // d_in[1] = causal mask (bool) — causality handled analytically
  const float* wq = (const float*)d_in[2];
  const float* wk = (const float*)d_in[3];
  const float* wv = (const float*)d_in[4];
  const float* wo = (const float*)d_in[5];

  char* ws = (char*)d_ws;
  const size_t XB   = (size_t)BSc * Dc * 2;   // 16 MB
  const size_t WT   = (size_t)Dc * Dc * 2;    //  8 MB each
  __bf16* xb  = (__bf16*)(ws);
  __bf16* wqT = (__bf16*)(ws + XB);
  __bf16* wkT = (__bf16*)(ws + XB + WT);
  __bf16* wvT = (__bf16*)(ws + XB + 2 * WT);
  __bf16* woT = (__bf16*)(ws + XB + 3 * WT);
  __bf16* Qb  = (__bf16*)(ws + XB + 4 * WT);
  __bf16* Kb  = (__bf16*)(ws + 2 * XB + 4 * WT);
  __bf16* VbT = (__bf16*)(ws + 3 * XB + 4 * WT);
  __bf16* Ab  = (__bf16*)(ws + 4 * XB + 4 * WT);

  cvt_bf16_kernel<<<4096, 256, 0, stream>>>(x, xb, BSc * Dc);
  dim3 tg(Dc / 32, Dc / 32);
  dim3 tb(32, 32);
  tcvt_kernel<<<tg, tb, 0, stream>>>(wq, wqT);
  tcvt_kernel<<<tg, tb, 0, stream>>>(wk, wkT);
  tcvt_kernel<<<tg, tb, 0, stream>>>(wv, wvT);
  tcvt_kernel<<<tg, tb, 0, stream>>>(wo, woT);

  qkv_rope_kernel<<<dim3(BSc / 128, Dc / 64, 3), 128, 0, stream>>>(
      xb, wqT, wkT, wvT, Qb, Kb, VbT);

  flash_attn_kernel<<<Bc * Hc * (Sc / 16), 32, 0, stream>>>(Qb, Kb, VbT, Ab);

  out_proj_kernel<<<dim3(BSc / 128, Dc / 64), 128, 0, stream>>>(Ab, woT,
                                                                (float*)d_out);
}
